// GMatrixFingerprintKDecision_43241730736524
// MI455X (gfx1250) — matrix operand order repaired and checked by reading, not compiled, not run on hardware
//
#include <hip/hip_runtime.h>

typedef __attribute__((ext_vector_type(16))) _Float16 v16h;
typedef __attribute__((ext_vector_type(8)))  float    v8f;

#define IDIM 219   // sum_{k=2..10} k(k+1)/2
#define BATCH 256
#define TLEN 4096

__device__ __attribute__((noinline)) void sort_desc(float* a, int n) {
    for (int i = 1; i < n; ++i) {
        float v = a[i];
        int j = i - 1;
        while (j >= 0 && a[j] < v) { a[j + 1] = a[j]; --j; }
        a[j + 1] = v;
    }
}

__device__ inline float gelu_exact(float x) {
    return 0.5f * x * (1.0f + erff(x * 0.70710678118654752440f));
}

// Per-block body for one (batch row, K). 8 waves split T; each wave softmaxes
// 32 tokens/chunk in registers (lane = token), stages the
// chunk in its private LDS slab as [class][token] f16, rebuilds the A operand
// (M=class, K=token) per the CDNA5 16-bit A layout, and accumulates
// G = P^T P with v_wmma_f32_16x16x32_f16. Wave-local s_wait_dscnt replaces
// block barriers in the hot loop (each wave owns its slab; LDS is in-order
// per wave). Cross-wave reduction via ds_add_f32, then lane 0 sorts + emits.
template <int K>
__device__ __forceinline__ void gram_body(const float* __restrict__ logits,
                                          float* __restrict__ feats) {
    constexpr int OFF  = (K - 1) * K * (K + 1) / 6 - 1;  // feature offset for this K
    constexpr int NOFF = K * (K - 1) / 2;

    __shared__ _Float16 stage[8][16][32];  // per-wave [class][token] slab
    __shared__ float    red[256];          // 16x16 cross-wave Gram reduction

    const int b    = blockIdx.x;
    const int tid  = threadIdx.x;
    const int w    = tid >> 5;
    const int lane = tid & 31;
    const int g    = lane >> 4;

    // zero the padded classes once (tokens reuse the same slots every chunk;
    // only this wave reads this slab, so no block barrier needed)
    for (int c = K; c < 16; ++c) stage[w][c][lane] = (_Float16)0.0f;
    red[tid] = 0.0f;
    asm volatile("" ::: "memory");

    v8f acc = {};
    const int tokPerWave = TLEN >> 3;  // 512
    const int base_t     = w * tokPerWave;
    const float* rowbase = logits + (size_t)b * TLEN * K;

    for (int chunk = 0; chunk < tokPerWave; chunk += 32) {
        const int t = base_t + chunk + lane;
        const float* p = rowbase + (size_t)t * K;

        float l[K];
#pragma unroll
        for (int c = 0; c < K; ++c) l[c] = p[c];
        // prefetch next chunk of this wave's token stream
        __builtin_prefetch((const void*)(p + 32 * K), 0, 1);

        float m = l[0];
#pragma unroll
        for (int c = 1; c < K; ++c) m = fmaxf(m, l[c]);
        float s = 0.0f;
#pragma unroll
        for (int c = 0; c < K; ++c) { l[c] = __expf(l[c] - m); s += l[c]; }
        const float inv = __builtin_amdgcn_rcpf(s);   // v_rcp_f32, not div chain

        // softmax probs for staging (lane = local token, element e = class):
        // entirely lane-local.
        union { v16h h; unsigned u[8]; } Bv;
#pragma unroll
        for (int e = 0; e < 16; ++e)
            Bv.h[e] = (e < K) ? (_Float16)(l[e] * inv) : (_Float16)0.0f;

        // stage this wave's chunk: [class][token] halves
#pragma unroll
        for (int c = 0; c < K; ++c) stage[w][c][lane] = Bv.h[c];

        // wave-local ordering only: stores complete before the transposed
        // reads below (LDS is in-order per wave; this also fences the
        // compiler from hoisting the loads)
        asm volatile("s_wait_dscnt 0x0" ::: "memory");

        // A operand (16x32, M=class=lane%16, K=token). Packed halves:
        // dword index within a class row = (v&3) + 8*(v>>2) + 4*(lane>>4)
        const unsigned* sp = (const unsigned*)&stage[w][0][0];
        const int cls16 = (lane & 15) * 16;
        union { v16h h; unsigned u[8]; } Av;
#pragma unroll
        for (int v = 0; v < 8; ++v)
            Av.u[v] = sp[cls16 + (v & 3) + ((v >> 2) << 3) + (g << 2)];

        acc = __builtin_amdgcn_wmma_f32_16x16x32_f16(
            /*neg_a=*/false, Av.h, /*neg_b=*/false, Av.h,
            /*c_mod=*/(short)0, acc, /*reuse_a=*/false, /*reuse_b=*/false);
    }

    __syncthreads();
    // C/D layout: lane L, vgpr v -> M = v + 8*(L/16), N = L%16
#pragma unroll
    for (int v = 0; v < 8; ++v) {
        const int m = v + (g << 3);
        const int n = lane & 15;
        atomicAdd(&red[m * 16 + n], acc[v]);
    }
    __syncthreads();

    if (tid == 0) {
        const float invT = 1.0f / (float)TLEN;
        float d[K];
        for (int i = 0; i < K; ++i) d[i] = red[i * 16 + i] * invT;
        sort_desc(d, K);
        float o[NOFF];
        int n = 0;
        for (int i = 0; i < K; ++i)
            for (int j = i + 1; j < K; ++j) o[n++] = red[i * 16 + j] * invT;
        sort_desc(o, NOFF);
        float* dst = feats + (size_t)b * IDIM + OFF;
        for (int i = 0; i < K; ++i) dst[i] = d[i];
        for (int i = 0; i < NOFF; ++i) dst[K + i] = o[i];
    }
}

struct PtrPack { const float* p[9]; };

// Single fused launch over all nine k-values: grid (BATCH, 9). blockIdx.y is
// block-uniform, so the switch costs one scalar branch per block and the
// whole 226 MB stream is in flight in one dispatch.
__global__ __launch_bounds__(256) void gram_all(PtrPack in, float* __restrict__ feats) {
    switch (blockIdx.y) {
        case 0: gram_body<2>(in.p[0], feats); break;
        case 1: gram_body<3>(in.p[1], feats); break;
        case 2: gram_body<4>(in.p[2], feats); break;
        case 3: gram_body<5>(in.p[3], feats); break;
        case 4: gram_body<6>(in.p[4], feats); break;
        case 5: gram_body<7>(in.p[5], feats); break;
        case 6: gram_body<8>(in.p[6], feats); break;
        case 7: gram_body<9>(in.p[7], feats); break;
        default: gram_body<10>(in.p[8], feats); break;
    }
}

// One block of 256 threads per batch row: LayerNorm over 219 feats, then the
// 219->256->256->9 GELU MLP (one thread per hidden unit). Trivial FLOPs.
__global__ __launch_bounds__(256) void mlp_kernel(
    const float* __restrict__ feats,
    const float* __restrict__ gamma, const float* __restrict__ beta,
    const float* __restrict__ w1, const float* __restrict__ b1,
    const float* __restrict__ w2, const float* __restrict__ b2,
    const float* __restrict__ w3, const float* __restrict__ b3,
    float* __restrict__ out) {
    __shared__ float x[IDIM];
    __shared__ float h[256];
    __shared__ float red[256];

    const int b = blockIdx.x;
    const int t = threadIdx.x;

    const float v = (t < IDIM) ? feats[(size_t)b * IDIM + t] : 0.0f;
    red[t] = v;
    __syncthreads();
    for (int s = 128; s > 0; s >>= 1) { if (t < s) red[t] += red[t + s]; __syncthreads(); }
    const float mu = red[0] * (1.0f / (float)IDIM);
    __syncthreads();

    const float dv = (t < IDIM) ? (v - mu) : 0.0f;
    red[t] = dv * dv;
    __syncthreads();
    for (int s = 128; s > 0; s >>= 1) { if (t < s) red[t] += red[t + s]; __syncthreads(); }
    const float var = red[0] * (1.0f / (float)IDIM);
    __syncthreads();

    if (t < IDIM) x[t] = dv * rsqrtf(var + 1e-5f) * gamma[t] + beta[t];
    __syncthreads();

    float a1 = b1[t];
    for (int i = 0; i < IDIM; ++i) a1 = fmaf(x[i], w1[i * 256 + t], a1);
    a1 = gelu_exact(a1);
    h[t] = a1;
    __syncthreads();

    float a2 = b2[t];
    for (int i = 0; i < 256; ++i) a2 = fmaf(h[i], w2[i * 256 + t], a2);
    a2 = gelu_exact(a2);
    __syncthreads();
    red[t] = a2;  // reuse as h2
    __syncthreads();

    if (t < 9) {
        float a3 = b3[t];
        for (int i = 0; i < 256; ++i) a3 = fmaf(red[i], w3[i * 9 + t], a3);
        out[b * 9 + t] = a3;
    }
}

extern "C" void kernel_launch(void* const* d_in, const int* in_sizes, int n_in,
                              void* d_out, int out_size, void* d_ws, size_t ws_size,
                              hipStream_t stream) {
    (void)in_sizes; (void)n_in; (void)out_size; (void)ws_size;
    float* feats = reinterpret_cast<float*>(d_ws);  // [256][219] f32

    PtrPack in;
    for (int i = 0; i < 9; ++i) in.p[i] = (const float*)d_in[i];

    gram_all<<<dim3(BATCH, 9, 1), 256, 0, stream>>>(in, feats);

    mlp_kernel<<<BATCH, 256, 0, stream>>>(
        feats,
        (const float*)d_in[9],  (const float*)d_in[10],
        (const float*)d_in[11], (const float*)d_in[12],
        (const float*)d_in[13], (const float*)d_in[14],
        (const float*)d_in[15], (const float*)d_in[16],
        (float*)d_out);
}
